// GroupQueryAttention_42133629174214
// MI455X (gfx1250) — matrix operand order repaired and checked by reading, not compiled, not run on hardware
//
#include <hip/hip_runtime.h>
#include <hip/hip_bf16.h>
#include <math.h>

typedef __attribute__((ext_vector_type(16))) __bf16 v16bf;
typedef __attribute__((ext_vector_type(8)))  float  v8f;

#define WMMA_BF16(A_, Bm_, C_) \
  __builtin_amdgcn_wmma_f32_16x16x32_bf16(false, (A_), false, (Bm_), (short)0, (C_), false, false)

constexpr int kB   = 2;
constexpr int kS   = 2048;
constexpr int kD   = 2048;     // d_in
constexpr int kNH  = 16;       // query heads
constexpr int kNKV = 4;        // kv heads
constexpr int kHD  = 128;      // head dim
constexpr int kM   = kB * kS;  // 4096 token rows

// ---------------------------------------------------------------------------
// CDNA5 async global->LDS copy (ASYNCcnt-tracked, no VGPR round trip).
// vdst = 32-bit LDS byte address, vaddr = 64-bit global address (GV mode).
// ---------------------------------------------------------------------------
__device__ __forceinline__ void async_copy_b128(const __bf16* lds_dst,
                                                const __bf16* gsrc) {
  uint32_t lds_addr = (uint32_t)(uintptr_t)lds_dst;   // low 32 bits = LDS offset
  uint64_t gaddr    = (uint64_t)(uintptr_t)gsrc;
  asm volatile("global_load_async_to_lds_b128 %0, %1, off"
               :: "v"(lds_addr), "v"(gaddr) : "memory");
}
__device__ __forceinline__ void wait_async0() {
  asm volatile("s_wait_asynccnt 0x0" ::: "memory");
}

// ---------------------------------------------------------------------------
// fp32 -> bf16 cast
// ---------------------------------------------------------------------------
__global__ __launch_bounds__(256) void f2bf_kernel(const float* __restrict__ src,
                                                   __bf16* __restrict__ dst, int n) {
  int i = blockIdx.x * 256 + threadIdx.x;
  if (i < n) dst[i] = (__bf16)src[i];
}

// ---------------------------------------------------------------------------
// Shared 128x128 tile GEMM mainloop: C[128,128] += A[128,K] * B[128,K]^T.
// Double-buffered LDS; tile k+1 prefetched with async-to-LDS copies while
// WMMAs consume tile k. 256 threads = 8 waves; wave w owns rows [w*16,w*16+16).
// lds_a / lds_b must each hold 2*128*32 bf16 (16 KB).
// ---------------------------------------------------------------------------
__device__ __forceinline__ void gemm_tile_128x128(
    const __bf16* __restrict__ A, int lda,
    const __bf16* __restrict__ B, int ldb,
    int K, __bf16* lds_a, __bf16* lds_b, v8f acc[8]) {
  const int tid  = threadIdx.x;
  const int lane = tid & 31;
  const int wave = tid >> 5;
  const int half = lane >> 4;
  const int ln   = lane & 15;
  v8f z = {};
  #pragma unroll
  for (int c = 0; c < 8; ++c) acc[c] = z;

  // Each thread owns two fixed 16-byte chunks per operand per k-step:
  // rows r0 and r0+64, columns [c0, c0+8).
  const int r0 = tid >> 2;
  const int c0 = (tid & 3) * 8;

  auto issue = [&](int k0, int buf) {
    const __bf16* la = lds_a + buf * 4096;
    const __bf16* lb = lds_b + buf * 4096;
    async_copy_b128(&la[r0 * 32 + c0],        &A[(size_t)r0 * lda + k0 + c0]);
    async_copy_b128(&la[(r0 + 64) * 32 + c0], &A[(size_t)(r0 + 64) * lda + k0 + c0]);
    async_copy_b128(&lb[r0 * 32 + c0],        &B[(size_t)r0 * ldb + k0 + c0]);
    async_copy_b128(&lb[(r0 + 64) * 32 + c0], &B[(size_t)(r0 + 64) * ldb + k0 + c0]);
  };

  int buf = 0;
  issue(0, 0);
  for (int k0 = 0; k0 < K; k0 += 32) {
    wait_async0();      // my copies into buf landed
    __syncthreads();    // everyone's copies landed; prior reads of buf^1 done
    if (k0 + 32 < K) issue(k0 + 32, buf ^ 1);

    const __bf16* la = lds_a + buf * 4096;
    const __bf16* lb = lds_b + buf * 4096;
    v16bf afrag = *(const v16bf*)&la[(wave * 16 + ln) * 32 + half * 16];
    #pragma unroll
    for (int g = 0; g < 2; ++g) {
      v16bf bfr[4];
      #pragma unroll
      for (int c = 0; c < 4; ++c)
        bfr[c] = *(const v16bf*)&lb[((g * 4 + c) * 16 + ln) * 32 + half * 16];
      #pragma unroll
      for (int c = 0; c < 4; ++c)
        acc[g * 4 + c] = WMMA_BF16(afrag, bfr[c], acc[g * 4 + c]);
    }
    buf ^= 1;
  }
}

// ---------------------------------------------------------------------------
// QKV projection + fused RMSNorm + RoPE epilogue.
// grid = (32 M-tiles, 24 head-tiles): nt<16 -> Q head, <20 -> K head, else V.
// Each wave's 16x128 strip is one head's full dim for 16 tokens, so RMSNorm
// (row reduction over 128 dims) and RoPE (d <-> d+64 pairing) stay in-register.
// ---------------------------------------------------------------------------
__global__ __launch_bounds__(256) void qkv_kernel(
    const __bf16* __restrict__ xb,
    const __bf16* __restrict__ wqb, const __bf16* __restrict__ wkb,
    const __bf16* __restrict__ wvb,
    const float* __restrict__ sin_t, const float* __restrict__ cos_t,
    const float* __restrict__ q_scale, const float* __restrict__ q_shift,
    const float* __restrict__ k_scale, const float* __restrict__ k_shift,
    __bf16* __restrict__ qb, __bf16* __restrict__ kb, __bf16* __restrict__ vb) {
  __shared__ __attribute__((aligned(32))) __bf16 lds_a[2 * 128 * 32];
  __shared__ __attribute__((aligned(32))) __bf16 lds_b[2 * 128 * 32];

  const int mt = blockIdx.x;   // 0..31
  const int nt = blockIdx.y;   // 0..23
  const __bf16* Bp;
  int kind, h;
  if (nt < 16)      { Bp = wqb + (size_t)nt * kHD * kD;        kind = 0; h = nt; }
  else if (nt < 20) { Bp = wkb + (size_t)(nt - 16) * kHD * kD; kind = 1; h = nt - 16; }
  else              { Bp = wvb + (size_t)(nt - 20) * kHD * kD; kind = 2; h = nt - 20; }

  v8f acc[8];
  gemm_tile_128x128(xb + (size_t)mt * 128 * kD, kD, Bp, kD, kD, lds_a, lds_b, acc);

  const int lane = threadIdx.x & 31, wave = threadIdx.x >> 5;
  const int half = lane >> 4, ln = lane & 15;
  const int m0 = mt * 128 + wave * 16;

  if (kind == 2) {  // V: plain bf16 scatter into [b, kv_head, s, d]
    #pragma unroll
    for (int c = 0; c < 8; ++c)
      #pragma unroll
      for (int r = 0; r < 8; ++r) {
        int m = m0 + r + half * 8, bi = m >> 11, tok = m & (kS - 1);
        vb[(((size_t)bi * kNKV + h) * kS + tok) * kHD + c * 16 + ln] = (__bf16)acc[c][r];
      }
    return;
  }

  const float* scl = (kind == 0) ? q_scale : k_scale;
  const float* shf = (kind == 0) ? q_shift : k_shift;

  // RMSNorm: per-row sum of squares over 128 dims (8 subtiles x 16 lanes).
  float ss[8];
  #pragma unroll
  for (int r = 0; r < 8; ++r) {
    float p = 0.f;
    #pragma unroll
    for (int c = 0; c < 8; ++c) p += acc[c][r] * acc[c][r];
    #pragma unroll
    for (int mm = 1; mm < 16; mm <<= 1) p += __shfl_xor(p, mm, 32);
    ss[r] = p;
  }
  #pragma unroll
  for (int c = 0; c < 8; ++c) {
    int d = c * 16 + ln;
    float g = 1.0f + scl[d], b0 = shf[d];
    #pragma unroll
    for (int r = 0; r < 8; ++r) {
      float inv = rsqrtf(ss[r] * (1.0f / 128.0f) + 1e-6f);
      acc[c][r] = acc[c][r] * inv * g + b0;
    }
  }

  // RoPE: pair d (subtile c) with d+64 (subtile c+4); scatter bf16 [b,h,s,d].
  __bf16* out = (kind == 0) ? qb : kb;
  const int nh = (kind == 0) ? kNH : kNKV;
  #pragma unroll
  for (int c = 0; c < 4; ++c) {
    int d = c * 16 + ln;
    #pragma unroll
    for (int r = 0; r < 8; ++r) {
      int m = m0 + r + half * 8, bi = m >> 11, tok = m & (kS - 1);
      float s1 = sin_t[tok * kHD + d],      c1 = cos_t[tok * kHD + d];
      float s2 = sin_t[tok * kHD + d + 64], c2 = cos_t[tok * kHD + d + 64];
      float x1 = acc[c][r], x2 = acc[c + 4][r];
      size_t base = (((size_t)bi * nh + h) * kS + tok) * kHD;
      out[base + d]      = (__bf16)(x1 * c1 - x2 * s1);
      out[base + d + 64] = (__bf16)(x2 * c2 + x1 * s2);
    }
  }
}

// ---------------------------------------------------------------------------
// Flash attention with sink key. One wave per 16-query tile; 128 threads/block
// (4 waves). Keys streamed in chunks of 32: QK^T = 2x4 WMMAs, online fp32
// softmax (running max seeded with the sink logit, l seeded with 1 so the sink
// weight drops out), P repacked via LDS, PV = 8 WMMAs.
// ---------------------------------------------------------------------------
__global__ __launch_bounds__(128) void attn_kernel(
    const __bf16* __restrict__ qb, const __bf16* __restrict__ kb,
    const __bf16* __restrict__ vb, const float* __restrict__ k_sink,
    __bf16* __restrict__ attnb) {
  __shared__ __attribute__((aligned(32))) __bf16 p_lds_all[4][16 * 32];

  const int lane = threadIdx.x & 31, wave = threadIdx.x >> 5;
  const int half = lane >> 4, ln = lane & 15;
  const int bh = blockIdx.x;             // bi*16 + h
  const int bi = bh >> 4, h = bh & 15;
  const int kvh = h >> 2;                // 4 query heads per kv head
  const int qt = blockIdx.y * 4 + wave;  // 0..127 query tiles

  const __bf16* qptr = qb + ((size_t)bi * kNH + h) * kS * kHD;
  const __bf16* kptr = kb + ((size_t)bi * kNKV + kvh) * kS * kHD;
  const __bf16* vptr = vb + ((size_t)bi * kNKV + kvh) * kS * kHD;
  __bf16* p_lds = p_lds_all[wave];
  const float sc = 0.08838834764831845f;  // 128^-0.5

  // Q A-fragments for this wave's 16 rows (4 K-chunks of 32 dims).
  v16bf aq[4];
  #pragma unroll
  for (int kk = 0; kk < 4; ++kk)
    aq[kk] = *(const v16bf*)&qptr[(size_t)(qt * 16 + ln) * kHD + kk * 32 + half * 16];

  v8f acc[8];
  v8f z = {};
  #pragma unroll
  for (int c = 0; c < 8; ++c) acc[c] = z;

  // Sink logit per row -> seeds running max; l=1 (its weight is then dropped).
  const float* sink = k_sink + h * kHD;
  float mrow[8], lrow[8];
  #pragma unroll
  for (int r = 0; r < 8; ++r) {
    int row = qt * 16 + r + half * 8;
    float p = 0.f;
    #pragma unroll
    for (int j = 0; j < 8; ++j) {
      int d = ln * 8 + j;
      p += (float)qptr[(size_t)row * kHD + d] * sink[d];
    }
    #pragma unroll
    for (int mm = 1; mm < 16; mm <<= 1) p += __shfl_xor(p, mm, 32);
    mrow[r] = p * sc;
    lrow[r] = 1.0f;
  }

  const int kend = qt * 16 + 16;  // causal: keys <= query
  for (int kt = 0; kt * 32 < kend; ++kt) {
    // Scores: 16 queries x 32 keys.
    v8f sco[2];
    #pragma unroll
    for (int n = 0; n < 2; ++n) {
      sco[n] = z;
      v16bf bk[4];
      #pragma unroll
      for (int kk = 0; kk < 4; ++kk) {
        int key = kt * 32 + n * 16 + ln;
        bk[kk] = *(const v16bf*)&kptr[(size_t)key * kHD + kk * 32 + half * 16];
      }
      #pragma unroll
      for (int kk = 0; kk < 4; ++kk) sco[n] = WMMA_BF16(aq[kk], bk[kk], sco[n]);
    }
    // Online softmax update + stash P (bf16) in LDS for the PV matmul.
    #pragma unroll
    for (int r = 0; r < 8; ++r) {
      int qrow = qt * 16 + r + half * 8;
      float mx = -INFINITY;
      #pragma unroll
      for (int n = 0; n < 2; ++n) {
        int key = kt * 32 + n * 16 + ln;
        float v = sco[n][r] * sc;
        if (key > qrow) v = -INFINITY;
        sco[n][r] = v;
        mx = fmaxf(mx, v);
      }
      #pragma unroll
      for (int mm = 1; mm < 16; mm <<= 1) mx = fmaxf(mx, __shfl_xor(mx, mm, 32));
      float mnew = fmaxf(mrow[r], mx);
      float corr = __expf(mrow[r] - mnew);
      mrow[r] = mnew;
      float rs = 0.f;
      #pragma unroll
      for (int n = 0; n < 2; ++n) {
        float p = __expf(sco[n][r] - mnew);  // exp(-inf)=0 handles the mask
        rs += p;
        p_lds[(r + half * 8) * 32 + n * 16 + ln] = (__bf16)p;
      }
      #pragma unroll
      for (int mm = 1; mm < 16; mm <<= 1) rs += __shfl_xor(rs, mm, 32);
      lrow[r] = lrow[r] * corr + rs;
      #pragma unroll
      for (int c = 0; c < 8; ++c) acc[c][r] = acc[c][r] * corr;
    }
    // PV: P[16x32] (A from LDS) x V[32x128] -> 8 WMMAs.
    v16bf ap = *(const v16bf*)&p_lds[ln * 32 + half * 16];
    #pragma unroll
    for (int c = 0; c < 8; ++c) {
      v16bf bv;
      #pragma unroll
      for (int s = 0; s < 16; ++s)
        bv[s] = vptr[(size_t)(kt * 32 + half * 16 + s) * kHD + c * 16 + ln];
      acc[c] = WMMA_BF16(ap, bv, acc[c]);
    }
  }

  // out = acc / l, stored bf16 in [token_row, h*128+d] (A layout for O-proj).
  #pragma unroll
  for (int r = 0; r < 8; ++r) {
    int row = qt * 16 + r + half * 8;
    float inv = 1.0f / lrow[r];
    #pragma unroll
    for (int c = 0; c < 8; ++c)
      attnb[((size_t)bi * kS + row) * kD + h * kHD + c * 16 + ln] =
          (__bf16)(acc[c][r] * inv);
  }
}

// ---------------------------------------------------------------------------
// Output projection: attn[4096,2048](bf16) @ wo[2048,2048]^T -> fp32 d_out.
// ---------------------------------------------------------------------------
__global__ __launch_bounds__(256) void oproj_kernel(
    const __bf16* __restrict__ attnb, const __bf16* __restrict__ wob,
    float* __restrict__ out) {
  __shared__ __attribute__((aligned(32))) __bf16 lds_a[2 * 128 * 32];
  __shared__ __attribute__((aligned(32))) __bf16 lds_b[2 * 128 * 32];
  const int mt = blockIdx.x, nt = blockIdx.y;
  v8f acc[8];
  gemm_tile_128x128(attnb + (size_t)mt * 128 * kD, kD,
                    wob + (size_t)nt * 128 * kD, kD, kD, lds_a, lds_b, acc);
  const int lane = threadIdx.x & 31, wave = threadIdx.x >> 5;
  const int half = lane >> 4, ln = lane & 15;
  const int m0 = mt * 128 + wave * 16;
  #pragma unroll
  for (int c = 0; c < 8; ++c)
    #pragma unroll
    for (int r = 0; r < 8; ++r)
      out[(size_t)(m0 + r + half * 8) * kD + nt * 128 + c * 16 + ln] = acc[c][r];
}

// ---------------------------------------------------------------------------
extern "C" void kernel_launch(void* const* d_in, const int* in_sizes, int n_in,
                              void* d_out, int out_size, void* d_ws, size_t ws_size,
                              hipStream_t stream) {
  (void)in_sizes; (void)n_in; (void)out_size; (void)ws_size;
  const float* x       = (const float*)d_in[0];
  // d_in[1] = mask (bool) unused: causal mask computed analytically.
  const float* sin_t   = (const float*)d_in[2];
  const float* cos_t   = (const float*)d_in[3];
  const float* wq      = (const float*)d_in[4];
  const float* wk      = (const float*)d_in[5];
  const float* wv      = (const float*)d_in[6];
  const float* wo      = (const float*)d_in[7];
  const float* q_scale = (const float*)d_in[8];
  const float* q_shift = (const float*)d_in[9];
  const float* k_scale = (const float*)d_in[10];
  const float* k_shift = (const float*)d_in[11];
  const float* k_sink  = (const float*)d_in[12];
  float* out = (float*)d_out;

  // Workspace carve (bf16 buffers), 256-B aligned, ~76 MB total.
  char* ws = (char*)d_ws;
  size_t off = 0;
  auto carve = [&](size_t elems) {
    void* p = ws + off;
    off += (elems * sizeof(__bf16) + 255) & ~(size_t)255;
    return (__bf16*)p;
  };
  __bf16* xb    = carve((size_t)kM * kD);               // 16 MB
  __bf16* wqb   = carve((size_t)kNH * kHD * kD);        //  8 MB
  __bf16* wkb   = carve((size_t)kNKV * kHD * kD);       //  2 MB
  __bf16* wvb   = carve((size_t)kNKV * kHD * kD);       //  2 MB
  __bf16* wob   = carve((size_t)kD * kD);               //  8 MB
  __bf16* qb    = carve((size_t)kB * kNH * kS * kHD);   // 16 MB
  __bf16* kbuf  = carve((size_t)kB * kNKV * kS * kHD);  //  4 MB
  __bf16* vbuf  = carve((size_t)kB * kNKV * kS * kHD);  //  4 MB
  __bf16* attnb = carve((size_t)kM * kD);               // 16 MB

  auto cast = [&](const float* s, __bf16* d, int n) {
    f2bf_kernel<<<(n + 255) / 256, 256, 0, stream>>>(s, d, n);
  };
  cast(x,  xb,  kM * kD);
  cast(wq, wqb, kNH * kHD * kD);
  cast(wk, wkb, kNKV * kHD * kD);
  cast(wv, wvb, kNKV * kHD * kD);
  cast(wo, wob, kD * kD);

  dim3 g1(kM / 128, kNH + 2 * kNKV);  // 32 x 24
  qkv_kernel<<<g1, 256, 0, stream>>>(xb, wqb, wkb, wvb, sin_t, cos_t,
                                     q_scale, q_shift, k_scale, k_shift,
                                     qb, kbuf, vbuf);

  dim3 g2(kB * kNH, (kS / 16) / 4);   // 32 x 32, 4 q-tiles (waves) per block
  attn_kernel<<<g2, 128, 0, stream>>>(qb, kbuf, vbuf, k_sink, attnb);

  dim3 g3(kM / 128, kD / 128);        // 32 x 16
  oproj_kernel<<<g3, 256, 0, stream>>>(attnb, wob, out);
}
